// AdvancedAnomalyGNN_46514495816138
// MI455X (gfx1250) — compile-verified, matrix operand order checked by reading, not run on hardware
//
#include <hip/hip_runtime.h>
#include <hip/hip_bf16.h>

// ---------------------------------------------------------------------------
// AdvancedAnomalyGNN: 2-layer GATv2 for MI455X (gfx1250, wave32, WMMA bf16)
// ---------------------------------------------------------------------------
// Dense transforms: one fused WMMA GEMM per layer, A read once per M-strip,
// tiles staged global->LDS by the Tensor Data Mover (tensor_load_to_lds,
// TENSORcnt) when available, then fp32->bf16 converted in LDS for
// v_wmma_f32_16x16x32_bf16.  Edge phase: 1 wave32 per edge, 8 lanes per head,
// float4 gathers, global_atomic_add_f32 scatters, ordered-uint atomicMax.
// ---------------------------------------------------------------------------

typedef __attribute__((ext_vector_type(16))) __bf16 v16bf;
typedef __attribute__((ext_vector_type(8)))  __bf16 v8bf;
typedef __attribute__((ext_vector_type(8)))  float  v8f;
typedef __attribute__((ext_vector_type(4)))  unsigned int u32x4;
typedef __attribute__((ext_vector_type(8)))  int i32x8;
typedef __attribute__((ext_vector_type(4)))  int i32x4;

#define D_IN 128
#define HID  32
#define OUTC 128
#define HD   128
#define NEG_SLOPE 0.2f
#define LN_EPS 1e-5f

#if __has_builtin(__builtin_amdgcn_tensor_load_to_lds) && \
    __has_builtin(__builtin_amdgcn_s_wait_tensorcnt)
#define USE_TDM 1
#else
#define USE_TDM 0
#endif

// ---------------- helpers ----------------
__device__ __forceinline__ unsigned short f2bf(float f) {
    unsigned int x = __float_as_uint(f);
    unsigned int r = x + 0x7FFFu + ((x >> 16) & 1u);   // round-to-nearest-even
    return (unsigned short)(r >> 16);
}
// monotone float->uint encoding: larger float => larger uint; 0u == minimum.
__device__ __forceinline__ unsigned int ford(float f) {
    unsigned int u = __float_as_uint(f);
    return (u & 0x80000000u) ? ~u : (u | 0x80000000u);
}
__device__ __forceinline__ float funord(unsigned int u) {
    return __uint_as_float((u & 0x80000000u) ? (u & 0x7FFFFFFFu) : ~u);
}

#if USE_TDM
// Issue one 2D-tile TDM load: global (row-strided, fp32) -> LDS (packed).
// D# layout per CDNA5 ISA ch.8: group0 {count, lds_addr, global_addr, type=2},
// group1 {data_size=4B, tensor_dim0/1, tile_dim0/1, tensor_dim0_stride}.
// This toolchain exposes the 6-argument builtin:
//   (u32x4 g0, i32x8 g1, i32x4 g2, i32x4 g3, i32x8 g4, i32 cpol)
__device__ __forceinline__ void tdm_load_2d(unsigned lds_off, const void* gptr,
                                            unsigned dim0, unsigned dim1,
                                            unsigned tile0, unsigned tile1,
                                            unsigned long long stride_elems) {
    unsigned long long ga = (unsigned long long)(uintptr_t)gptr;
    u32x4 g0;
    g0[0] = 1u;                                         // count=1, user desc
    g0[1] = lds_off;                                    // lds_addr (bytes)
    g0[2] = (unsigned)(ga & 0xFFFFFFFFu);               // global_addr lo
    g0[3] = (unsigned)((ga >> 32) & 0x1FFFFFFu) | (2u << 30);  // hi | type=2
    i32x8 g1;
    g1[0] = (int)(2u << 16);                            // data_size = 4 bytes
    g1[1] = (int)((dim0 & 0xFFFFu) << 16);              // tensor_dim0 lo
    g1[2] = (int)((dim0 >> 16) | ((dim1 & 0xFFFFu) << 16));
    g1[3] = (int)((dim1 >> 16) | ((tile0 & 0xFFFFu) << 16));
    g1[4] = (int)(tile1 & 0xFFFFu);                     // tile_dim1, dim2=0
    g1[5] = (int)(unsigned)(stride_elems & 0xFFFFFFFFu);
    g1[6] = (int)(unsigned)((stride_elems >> 32) & 0xFFFFu);
    g1[7] = 0;
    i32x4 z4 = {0, 0, 0, 0};
    i32x8 z8 = {0, 0, 0, 0, 0, 0, 0, 0};
    __builtin_amdgcn_tensor_load_to_lds(g0, g1, z4, z4, z8, 0);
}
#endif

// ---------------- mean(edge_weight) ----------------
__global__ void k_reduce_sum(const float* __restrict__ x, int n, float* out) {
    float s = 0.f;
    for (int i = blockIdx.x * blockDim.x + threadIdx.x; i < n;
         i += gridDim.x * blockDim.x) s += x[i];
    #pragma unroll
    for (int m = 1; m < 32; m <<= 1) s += __shfl_xor(s, m, 32);
    if ((threadIdx.x & 31) == 0) atomicAdd(out, s);
}

// ------------- fused WMMA bf16 GEMM: [C0|C1|C2] = A * [B0|B1|B2] -------------
// Block tile 128(M) x 64(N), K-step 32; 256 threads = 8 waves, each wave owns
// a 16-row slice and 4 n-tiles (4 v_wmma per K-step).
#define BM 128
#define BN 64
#define BK 32
#define APITCH 40   // ushort pitch (80B): conflict-free, 16B-aligned rows

__global__ __launch_bounds__(256)
void k_gemm_fused(const float* __restrict__ A, int M, int K,
                  const float* __restrict__ B0, float* __restrict__ C0, int n0,
                  const float* __restrict__ B1, float* __restrict__ C1, int n1,
                  const float* __restrict__ B2, float* __restrict__ C2, int n2) {
    __shared__ unsigned short As[BM * APITCH];
    __shared__ unsigned short Bs[BN * APITCH];
#if USE_TDM
    __shared__ float Afs[BM * BK];   // TDM staging (fp32)
    __shared__ float Bfs[BK * BN];
#endif

    const int tid  = threadIdx.x;
    const int wid  = tid >> 5;
    const int lane = tid & 31;
    const int hi   = lane >> 4;        // 0: K 0..7,16..23  1: K 8..15,24..31
    const int r    = lane & 15;
    const int bm   = blockIdx.y * BM;

    // segment select within fused [B0|B1|B2]
    int bnF = blockIdx.x * BN;
    const float* B; float* C; int Nt; int bn;
    if (bnF < n0)           { B = B0; C = C0; Nt = n0; bn = bnF; }
    else if (bnF < n0 + n1) { B = B1; C = C1; Nt = n1; bn = bnF - n0; }
    else                    { B = B2; C = C2; Nt = n2; bn = bnF - n0 - n1; }

    const int rows_rem = (M - bm) < BM ? (M - bm) : BM;

    v8f acc[4];
    #pragma unroll
    for (int t = 0; t < 4; ++t) acc[t] = (v8f)0.f;

    for (int kb = 0; kb < K; kb += BK) {
#if USE_TDM
        // ---- DMA fp32 tiles into LDS via Tensor Data Mover (wave 0 only;
        //      OOB rows beyond tensor_dim1 are zero-filled by hardware) ----
        if (wid == 0) {
            tdm_load_2d((unsigned)(uintptr_t)&Afs[0], A + (size_t)bm * K + kb,
                        BK, (unsigned)rows_rem, BK, BM, (unsigned long long)K);
            tdm_load_2d((unsigned)(uintptr_t)&Bfs[0], B + (size_t)kb * Nt + bn,
                        BN, BK, BN, BK, (unsigned long long)Nt);
            __builtin_amdgcn_s_wait_tensorcnt(0);
        }
        __syncthreads();
        // ---- convert fp32 staging -> padded bf16 operand tiles ----
        #pragma unroll
        for (int j = 0; j < 4; ++j) {               // A: 1024 float4
            int idx = tid + j * 256;
            int row = idx >> 3, c4 = idx & 7;
            float4 v = *reinterpret_cast<const float4*>(&Afs[row * BK + c4 * 4]);
            ushort4 b4;
            b4.x = f2bf(v.x); b4.y = f2bf(v.y); b4.z = f2bf(v.z); b4.w = f2bf(v.w);
            *reinterpret_cast<ushort4*>(&As[row * APITCH + c4 * 4]) = b4;
        }
        #pragma unroll
        for (int j = 0; j < 8; ++j) {               // B: transpose to [n][k]
            int idx = tid + j * 256;
            int k = idx >> 6, n = idx & 63;
            Bs[n * APITCH + k] = f2bf(Bfs[k * BN + n]);
        }
        __syncthreads();
#else
        // ---- fallback: direct global staging with bf16 convert ----
        #pragma unroll
        for (int j = 0; j < 4; ++j) {
            int idx = tid + j * 256;
            int row = idx >> 3, c4 = idx & 7;
            float4 v = make_float4(0.f, 0.f, 0.f, 0.f);
            if (row < rows_rem)
                v = *reinterpret_cast<const float4*>(
                        A + (size_t)(bm + row) * K + kb + c4 * 4);
            ushort4 b4;
            b4.x = f2bf(v.x); b4.y = f2bf(v.y); b4.z = f2bf(v.z); b4.w = f2bf(v.w);
            *reinterpret_cast<ushort4*>(&As[row * APITCH + c4 * 4]) = b4;
        }
        #pragma unroll
        for (int j = 0; j < 8; ++j) {
            int idx = tid + j * 256;
            int k = idx >> 6, n = idx & 63;
            Bs[n * APITCH + k] = f2bf(B[(size_t)(kb + k) * Nt + bn + n]);
        }
        if (kb + BK < K && (tid >> 3) < rows_rem)
            __builtin_prefetch(A + (size_t)(bm + (tid >> 3)) * K + kb + BK, 0, 1);
        __syncthreads();
#endif
        // ---- A fragment (16x32 bf16, ISA 7.12.2 lane layout) ----
        const unsigned short* arow = &As[(wid * 16 + r) * APITCH];
        v8bf a0 = *reinterpret_cast<const v8bf*>(arow + (hi ? 8 : 0));
        v8bf a1 = *reinterpret_cast<const v8bf*>(arow + 16 + (hi ? 8 : 0));
        v16bf af;
        #pragma unroll
        for (int i = 0; i < 8; ++i) { af[i] = a0[i]; af[i + 8] = a1[i]; }

        #pragma unroll
        for (int t = 0; t < 4; ++t) {
            const unsigned short* brow = &Bs[(t * 16 + r) * APITCH];
            v8bf b0 = *reinterpret_cast<const v8bf*>(brow + (hi ? 8 : 0));
            v8bf b1 = *reinterpret_cast<const v8bf*>(brow + 16 + (hi ? 8 : 0));
            v16bf bfr;
            #pragma unroll
            for (int i = 0; i < 8; ++i) { bfr[i] = b0[i]; bfr[i + 8] = b1[i]; }
            acc[t] = __builtin_amdgcn_wmma_f32_16x16x32_bf16(
                false, af, false, bfr, (short)0, acc[t], false, false);
        }
        __syncthreads();   // protect operand/staging LDS before next K-step
    }

    // ---- store C (VGPR j -> row wid*16 + j + 8*hi, col = lane&15) ----
    #pragma unroll
    for (int t = 0; t < 4; ++t) {
        int col = bn + t * 16 + r;
        #pragma unroll
        for (int j = 0; j < 8; ++j) {
            int row = wid * 16 + j + (hi ? 8 : 0);
            if (row < rows_rem) C[(size_t)(bm + row) * Nt + col] = acc[t][j];
        }
    }
}

// ---------------- edge scores: alpha[e,h] + segment max ----------------
template <int C>
__global__ void k_edge_scores(const float* __restrict__ xl,
                              const float* __restrict__ xr,
                              const float* __restrict__ We,
                              const float* __restrict__ att,
                              const float* __restrict__ ew,
                              const float* __restrict__ ew_sum,
                              const int* __restrict__ src_i,
                              const int* __restrict__ dst_i,
                              float* __restrict__ alpha,
                              unsigned int* __restrict__ amax,
                              int E, int N) {
    const int HC  = 4 * C;
    const int PER = C / 8;
    int e = blockIdx.x * 8 + (threadIdx.x >> 5);
    if (e >= E + N) return;
    int lane = threadIdx.x & 31;
    int s, d; float w;
    if (e < E) { s = src_i[e]; d = dst_i[e]; w = ew[e]; }
    else       { s = d = e - E; w = ew_sum[0] / (float)E; }
    int head = lane >> 3;
    const float* XL = xl + (size_t)s * HC;
    const float* XR = xr + (size_t)d * HC;
    int base = lane * PER;
    float acc = 0.f;
    #pragma unroll
    for (int q = 0; q < PER / 4; ++q) {
        float4 a = *reinterpret_cast<const float4*>(XL + base + 4 * q);
        float4 b = *reinterpret_cast<const float4*>(XR + base + 4 * q);
        float4 we = *reinterpret_cast<const float4*>(We + base + 4 * q);
        float4 at = *reinterpret_cast<const float4*>(att + base + 4 * q);
        float m;
        m = a.x + b.x + w * we.x; m = m > 0.f ? m : NEG_SLOPE * m; acc += m * at.x;
        m = a.y + b.y + w * we.y; m = m > 0.f ? m : NEG_SLOPE * m; acc += m * at.y;
        m = a.z + b.z + w * we.z; m = m > 0.f ? m : NEG_SLOPE * m; acc += m * at.z;
        m = a.w + b.w + w * we.w; m = m > 0.f ? m : NEG_SLOPE * m; acc += m * at.w;
    }
    acc += __shfl_xor(acc, 1, 32);
    acc += __shfl_xor(acc, 2, 32);
    acc += __shfl_xor(acc, 4, 32);
    if ((lane & 7) == 0) {
        alpha[(size_t)e * 4 + head] = acc;
        atomicMax(&amax[(size_t)d * 4 + head], ford(acc));
    }
}

// ---------------- softmax numerator + denominator ----------------
__global__ void k_edge_softmax(float* __restrict__ p,          // alpha -> exp
                               const int* __restrict__ dst_i,
                               const unsigned int* __restrict__ amax,
                               float* __restrict__ denom, int E, int N) {
    int i = blockIdx.x * blockDim.x + threadIdx.x;
    if (i >= (E + N) * 4) return;
    int e = i >> 2, h = i & 3;
    int d = (e < E) ? dst_i[e] : (e - E);
    float v = __expf(p[i] - funord(amax[(size_t)d * 4 + h]));
    p[i] = v;
    atomicAdd(&denom[(size_t)d * 4 + h], v);
}

// ---------------- scatter: agg[dst] += (p/denom) * xl[src] ----------------
template <int C>
__global__ void k_edge_scatter(const float* __restrict__ xl,
                               const float* __restrict__ p,
                               const float* __restrict__ denom,
                               const int* __restrict__ src_i,
                               const int* __restrict__ dst_i,
                               float* __restrict__ agg, int E, int N) {
    const int HC  = 4 * C;
    const int PER = C / 8;
    int e = blockIdx.x * 8 + (threadIdx.x >> 5);
    if (e >= E + N) return;
    int lane = threadIdx.x & 31;
    int s, d;
    if (e < E) { s = src_i[e]; d = dst_i[e]; } else { s = d = e - E; }
    int head = lane >> 3;
    float coef = p[(size_t)e * 4 + head] /
                 (denom[(size_t)d * 4 + head] + 1e-16f);
    const float* XL = xl + (size_t)s * HC;
    float* O = agg + (size_t)d * HC;
    int base = lane * PER;
    #pragma unroll
    for (int q = 0; q < PER / 4; ++q) {
        float4 a = *reinterpret_cast<const float4*>(XL + base + 4 * q);
        atomicAdd(&O[base + 4 * q + 0], coef * a.x);
        atomicAdd(&O[base + 4 * q + 1], coef * a.y);
        atomicAdd(&O[base + 4 * q + 2], coef * a.z);
        atomicAdd(&O[base + 4 * q + 3], coef * a.w);
    }
}

// ---------------- node epilogues: bias + ELU + LN + residual ----------------
template <int MEAN_HEADS>
__global__ void k_epilogue(const float* __restrict__ agg,
                           const float* __restrict__ bias,
                           const float* __restrict__ lng,
                           const float* __restrict__ lnb,
                           const float* __restrict__ resid,
                           const float* __restrict__ rbias,
                           float* __restrict__ out, int N) {
    int n = blockIdx.x, t = threadIdx.x;          // 128 threads
    float v;
    if (MEAN_HEADS) {
        const float* a = agg + (size_t)n * 512;
        v = 0.25f * (a[t] + a[128 + t] + a[256 + t] + a[384 + t]);
    } else {
        v = agg[(size_t)n * 128 + t];
    }
    v += bias[t];
    v = v > 0.f ? v : (__expf(v) - 1.f);          // ELU
    __shared__ float red[4];
    float s = v;
    #pragma unroll
    for (int m = 1; m < 32; m <<= 1) s += __shfl_xor(s, m, 32);
    if ((t & 31) == 0) red[t >> 5] = s;
    __syncthreads();
    float mean = (red[0] + red[1] + red[2] + red[3]) * (1.f / 128.f);
    __syncthreads();
    float dv = v - mean, sq = dv * dv;
    #pragma unroll
    for (int m = 1; m < 32; m <<= 1) sq += __shfl_xor(sq, m, 32);
    if ((t & 31) == 0) red[t >> 5] = sq;
    __syncthreads();
    float var = (red[0] + red[1] + red[2] + red[3]) * (1.f / 128.f);
    float o = dv * rsqrtf(var + LN_EPS) * lng[t] + lnb[t];
    out[(size_t)n * 128 + t] = o + resid[(size_t)n * 128 + t] + rbias[t];
}

// ---------------------------------------------------------------------------
extern "C" void kernel_launch(void* const* d_in, const int* in_sizes, int n_in,
                              void* d_out, int out_size, void* d_ws, size_t ws_size,
                              hipStream_t stream) {
    const float* x    = (const float*)d_in[0];
    const float* ew   = (const float*)d_in[1];
    const float* Wl0  = (const float*)d_in[2];
    const float* Wr0  = (const float*)d_in[3];
    const float* att0 = (const float*)d_in[4];
    const float* We0  = (const float*)d_in[5];
    const float* b0   = (const float*)d_in[6];
    const float* lng0 = (const float*)d_in[7];
    const float* lnb0 = (const float*)d_in[8];
    const float* rW0  = (const float*)d_in[9];
    const float* rb0  = (const float*)d_in[10];
    const float* Wl1  = (const float*)d_in[11];
    const float* Wr1  = (const float*)d_in[12];
    const float* att1 = (const float*)d_in[13];
    const float* We1  = (const float*)d_in[14];
    const float* b1   = (const float*)d_in[15];
    const float* lng1 = (const float*)d_in[16];
    const float* lnb1 = (const float*)d_in[17];
    const float* rW1  = (const float*)d_in[18];
    const float* rb1  = (const float*)d_in[19];
    const int*   ei   = (const int*)d_in[20];

    const int N = in_sizes[0] / D_IN;
    const int E = in_sizes[1];
    const int* src_i = ei;
    const int* dst_i = ei + E;
    const size_t Nn = (size_t)N;

    // ---- workspace layout (floats) ----
    float* ws    = (float*)d_ws;
    float* s_sum = ws;                       // 64 (padded)
    float* XL    = ws + 64;                  // N*512
    float* XR    = XL + Nn * 512;            // N*512
    float* AGG   = XR + Nn * 512;            // N*512
    float* RB    = AGG + Nn * 512;           // N*128
    float* HB    = RB + Nn * 128;            // N*128
    float* PB    = HB + Nn * 128;            // (E+N)*4
    unsigned int* AMAX = (unsigned int*)(PB + (size_t)(E + N) * 4); // N*4
    float* DEN   = (float*)(AMAX + Nn * 4);  // N*4

    const int ET = E + N;
    dim3 blk256(256), blk128(128);
    int edgeBlocks = (ET + 7) / 8;           // 8 waves / block
    int smBlocks   = (ET * 4 + 255) / 256;
    int mBlocks    = (N + BM - 1) / BM;

    // ---- mean edge weight ----
    (void)hipMemsetAsync(s_sum, 0, sizeof(float), stream);
    k_reduce_sum<<<512, blk256, 0, stream>>>(ew, E, s_sum);

    // ============================ layer 0 ============================
    (void)hipMemsetAsync(AMAX, 0, Nn * 4 * sizeof(unsigned int), stream);
    (void)hipMemsetAsync(DEN, 0, Nn * 4 * sizeof(float), stream);
    (void)hipMemsetAsync(AGG, 0, Nn * 128 * sizeof(float), stream);

    dim3 g0((128 + 128 + 128) / BN, mBlocks);   // fused [Wl0|Wr0|rW0]
    k_gemm_fused<<<g0, blk256, 0, stream>>>(x, N, D_IN,
                                            Wl0, XL, 128,
                                            Wr0, XR, 128,
                                            rW0, RB, 128);

    k_edge_scores<HID><<<edgeBlocks, blk256, 0, stream>>>(
        XL, XR, We0, att0, ew, s_sum, src_i, dst_i, PB, AMAX, E, N);
    k_edge_softmax<<<smBlocks, blk256, 0, stream>>>(PB, dst_i, AMAX, DEN, E, N);
    k_edge_scatter<HID><<<edgeBlocks, blk256, 0, stream>>>(
        XL, PB, DEN, src_i, dst_i, AGG, E, N);
    k_epilogue<0><<<N, blk128, 0, stream>>>(AGG, b0, lng0, lnb0, RB, rb0, HB, N);

    // ============================ layer 1 ============================
    (void)hipMemsetAsync(AMAX, 0, Nn * 4 * sizeof(unsigned int), stream);
    (void)hipMemsetAsync(DEN, 0, Nn * 4 * sizeof(float), stream);
    (void)hipMemsetAsync(AGG, 0, Nn * 512 * sizeof(float), stream);

    dim3 g1((512 + 512 + 128) / BN, mBlocks);   // fused [Wl1|Wr1|rW1]
    k_gemm_fused<<<g1, blk256, 0, stream>>>(HB, N, HD,
                                            Wl1, XL, 512,
                                            Wr1, XR, 512,
                                            rW1, RB, 128);

    k_edge_scores<OUTC><<<edgeBlocks, blk256, 0, stream>>>(
        XL, XR, We1, att1, ew, s_sum, src_i, dst_i, PB, AMAX, E, N);
    k_edge_softmax<<<smBlocks, blk256, 0, stream>>>(PB, dst_i, AMAX, DEN, E, N);
    k_edge_scatter<OUTC><<<edgeBlocks, blk256, 0, stream>>>(
        XL, PB, DEN, src_i, dst_i, AGG, E, N);
    k_epilogue<1><<<N, blk128, 0, stream>>>(AGG, b1, lng1, lnb1, RB, rb1,
                                            (float*)d_out, N);
}
